// FusedAttention_56221121904704
// MI455X (gfx1250) — matrix operand order repaired and checked by reading, not compiled, not run on hardware
//
#include <hip/hip_runtime.h>
#include <hip/hip_bf16.h>

// ---------------------------------------------------------------------------
// Fused causal multi-head attention for MI455X (gfx1250, wave32, WMMA).
//   B=4, T=2048, H=1024, NH=16, HD=64
// Pipeline: [1] qkv = x @ Wqkv^T (bf16 WMMA, fp32 acc) -> bf16 ws [3][B][NH][T][HD]
//           [2] flash attention per (b,h,64-row q block)  -> bf16 ws [B*T][H]
//           [3] out = attn @ Wout^T (bf16 WMMA, fp32 acc) -> fp32 d_out
// ---------------------------------------------------------------------------

typedef __bf16 bf16;
typedef __attribute__((ext_vector_type(8)))  __bf16 bf16x8;
typedef __attribute__((ext_vector_type(16))) __bf16 bf16x16;
typedef __attribute__((ext_vector_type(8)))  float  v8f;

#define B_   4
#define T_   2048
#define H_   1024
#define NH_  16
#define HD_  64
#define M_   (B_*T_)     // 8192
#define KDIM H_          // 1024

#define ASTR 40          // 128x32 bf16 A tile, 16B-aligned rows
#define BSTR 136         // 32x128 bf16 Bt tile, 16B-aligned rows
#define KSTR 72          // 64x64 attention tiles, 16B-aligned rows

__device__ __forceinline__ v8f wmma_bf16(bf16x16 a, bf16x16 b, v8f c) {
  // v_wmma_f32_16x16x32_bf16: (neg_a, A, neg_b, B, c_mod, C, reuse_a, reuse_b)
  return __builtin_amdgcn_wmma_f32_16x16x32_bf16(false, a, false, b, (short)0, c,
                                                 false, false);
}

__device__ __forceinline__ bf16x16 mk16(bf16x8 lo, bf16x8 hi) {
  bf16x16 r;
#pragma unroll
  for (int i = 0; i < 8; ++i) { r[i] = lo[i]; r[i + 8] = hi[i]; }
  return r;
}

// ---------------------------------------------------------------------------
// Kernel 1: qkv[p][b][h][t][d] (bf16) = x @ Wqkv^T ; 128x128x32 tiles, 8 waves
// ---------------------------------------------------------------------------
__global__ void __launch_bounds__(256)
qkv_gemm_kernel(const float* __restrict__ X, const float* __restrict__ W,
                bf16* __restrict__ qkv) {
  __shared__ bf16 As[128 * ASTR];  // A tile, row-major [m][k]
  __shared__ bf16 Bt[32 * BSTR];   // B tile, transposed [k][n]

  const int tid  = threadIdx.x;
  const int lane = tid & 31;
  const int wave = tid >> 5;
  const int wm   = wave >> 1;      // 0..3  (32-row slice)
  const int wn   = wave & 1;       // 0..1  (64-col slice)
  const int hi   = lane >> 4;      // half-wave: selects K sub-chunk
  const int ln   = lane & 15;

  const int m0 = blockIdx.y * 128;
  const int n0 = blockIdx.x * 128;

  v8f acc[2][4];
#pragma unroll
  for (int i = 0; i < 2; ++i)
#pragma unroll
    for (int j = 0; j < 4; ++j) acc[i][j] = (v8f)0.0f;

  for (int kb = 0; kb < KDIM; kb += 32) {
    __syncthreads();
    // X tile: 128x32 fp32 -> bf16 LDS
#pragma unroll
    for (int p = 0; p < 4; ++p) {
      int r = (tid >> 3) + p * 32;
      int c = (tid & 7) * 4;
      float4 f = *(const float4*)(X + (size_t)(m0 + r) * KDIM + kb + c);
      bf16* dst = &As[r * ASTR + c];
      dst[0] = (bf16)f.x; dst[1] = (bf16)f.y; dst[2] = (bf16)f.z; dst[3] = (bf16)f.w;
    }
    // W tile: 128(n) x 32(k) fp32 -> transposed bf16 Bt[k][n]
#pragma unroll
    for (int p = 0; p < 4; ++p) {
      int r = (tid >> 3) + p * 32;  // n
      int c = (tid & 7) * 4;        // k
      float4 f = *(const float4*)(W + (size_t)(n0 + r) * KDIM + kb + c);
      Bt[(c + 0) * BSTR + r] = (bf16)f.x;
      Bt[(c + 1) * BSTR + r] = (bf16)f.y;
      Bt[(c + 2) * BSTR + r] = (bf16)f.z;
      Bt[(c + 3) * BSTR + r] = (bf16)f.w;
    }
    if (kb + 32 < KDIM) {  // global_prefetch_b8 of next tiles
      __builtin_prefetch(X + (size_t)(m0 + (tid >> 3)) * KDIM + kb + 32, 0, 1);
      __builtin_prefetch(W + (size_t)(n0 + (tid >> 3)) * KDIM + kb + 32, 0, 1);
    }
    __syncthreads();

    bf16x16 afrag[2], bfrag[4];
#pragma unroll
    for (int mi = 0; mi < 2; ++mi) {
      int row = wm * 32 + mi * 16 + ln;  // A: lane<->M, regs<->K (split-K halves)
      bf16x8 lo = *(const bf16x8*)&As[row * ASTR + hi * 8];
      bf16x8 h8 = *(const bf16x8*)&As[row * ASTR + 16 + hi * 8];
      afrag[mi] = mk16(lo, h8);
    }
#pragma unroll
    for (int ni = 0; ni < 4; ++ni) {
      int col = wn * 64 + ni * 16 + ln;  // B: lane<->N, regs<->K (split-K halves)
#pragma unroll
      for (int e = 0; e < 16; ++e)
        bfrag[ni][e] = Bt[(((e & 8) << 1) + hi * 8 + (e & 7)) * BSTR + col];
    }
#pragma unroll
    for (int mi = 0; mi < 2; ++mi)
#pragma unroll
      for (int ni = 0; ni < 4; ++ni)
        acc[mi][ni] = wmma_bf16(afrag[mi], bfrag[ni], acc[mi][ni]);
  }

  // Epilogue: scatter bf16 into qkv[p][b][h][t][d]
#pragma unroll
  for (int mi = 0; mi < 2; ++mi)
#pragma unroll
    for (int ni = 0; ni < 4; ++ni)
#pragma unroll
      for (int v = 0; v < 8; ++v) {
        int m = m0 + wm * 32 + mi * 16 + v + 8 * hi;  // D: regs<->M halves
        int n = n0 + wn * 64 + ni * 16 + ln;          // D: lane<->N
        int p   = n >> 10;
        int rem = n & 1023;
        int h = rem >> 6, d = rem & 63;
        int b = m >> 11, t = m & 2047;
        size_t idx = ((((size_t)p * B_ + b) * NH_ + h) * T_ + t) * HD_ + d;
        qkv[idx] = (bf16)acc[mi][ni][v];
      }
}

// ---------------------------------------------------------------------------
// Kernel 2: flash attention. 4 waves, 64 q-rows per block, online softmax.
// ---------------------------------------------------------------------------
__global__ void __launch_bounds__(128)
attn_kernel(const bf16* __restrict__ qkv, bf16* __restrict__ aout) {
  __shared__ bf16 Kt[64 * KSTR];  // K transposed: [hd][key]
  __shared__ bf16 Vs[64 * KSTR];  // V as-is:     [key][hd]
  __shared__ bf16 Ps[64 * KSTR];  // P staging (wave-private 16-row slices)

  const int tid  = threadIdx.x;
  const int lane = tid & 31;
  const int wave = tid >> 5;
  const int hi   = lane >> 4;
  const int ln   = lane & 15;
  const int qb   = blockIdx.x;       // 0..31
  const int b    = blockIdx.y >> 4;
  const int h    = blockIdx.y & 15;

  const size_t head_off = (((size_t)b * NH_ + h) * T_) * HD_;
  const size_t part     = (size_t)B_ * NH_ * T_ * HD_;
  const bf16* Q = qkv + head_off;
  const bf16* K = qkv + part + head_off;
  const bf16* V = qkv + 2 * part + head_off;

  // Q fragments (held in registers for the whole kernel): A-layout
  bf16x16 qf[2];
  {
    int t = qb * 64 + wave * 16 + ln;
#pragma unroll
    for (int kc = 0; kc < 2; ++kc) {
      bf16x8 lo = *(const bf16x8*)(Q + (size_t)t * HD_ + kc * 32 + hi * 8);
      bf16x8 h8 = *(const bf16x8*)(Q + (size_t)t * HD_ + kc * 32 + 16 + hi * 8);
      qf[kc] = mk16(lo, h8);
    }
  }

  v8f o[4];
  float mrun[8], lrun[8];
#pragma unroll
  for (int i = 0; i < 4; ++i) o[i] = (v8f)0.0f;
#pragma unroll
  for (int v = 0; v < 8; ++v) { mrun[v] = -3.0e38f; lrun[v] = 0.0f; }

  const float scale = 0.125f;  // 1/sqrt(64)

  for (int j = 0; j <= qb; ++j) {
    __syncthreads();
    // Load K (transposed) and V (straight) 64x64 blocks into LDS
#pragma unroll
    for (int p = 0; p < 4; ++p) {
      int r = (tid >> 3) + p * 16;  // key row
      int c = (tid & 7) * 8;        // hd
      bf16x8 kv = *(const bf16x8*)(K + (size_t)(j * 64 + r) * HD_ + c);
#pragma unroll
      for (int e = 0; e < 8; ++e) Kt[(c + e) * KSTR + r] = kv[e];
      *(bf16x8*)&Vs[r * KSTR + c] =
          *(const bf16x8*)(V + (size_t)(j * 64 + r) * HD_ + c);
    }
    __syncthreads();

    // S = Q K^T  (per wave: 16 q-rows x 64 keys)
    v8f s[4];
#pragma unroll
    for (int ni = 0; ni < 4; ++ni) s[ni] = (v8f)0.0f;
#pragma unroll
    for (int kc = 0; kc < 2; ++kc) {
      int krow = kc * 32 + hi * 8;  // B: lane<->N(key), regs<->K(hd)
#pragma unroll
      for (int ni = 0; ni < 4; ++ni) {
        bf16x16 kf;
#pragma unroll
        for (int e = 0; e < 16; ++e)
          kf[e] = Kt[(krow + ((e & 8) << 1) + (e & 7)) * KSTR + ni * 16 + ln];
        s[ni] = wmma_bf16(qf[kc], kf, s[ni]);
      }
    }

    // Scale + causal mask (module uses -1e9, not -inf)
    float pv[4][8];
#pragma unroll
    for (int ni = 0; ni < 4; ++ni) {
      int kg = j * 64 + ni * 16 + ln;
#pragma unroll
      for (int v = 0; v < 8; ++v) {
        int qg = qb * 64 + wave * 16 + v + 8 * hi;
        float sv = s[ni][v] * scale;
        pv[ni][v] = (kg > qg) ? -1.0e9f : sv;
      }
    }

    // Online softmax: C/D layout puts one q-row across a 16-lane group
#pragma unroll
    for (int v = 0; v < 8; ++v) {
      float mx = pv[0][v];
#pragma unroll
      for (int ni = 1; ni < 4; ++ni) mx = fmaxf(mx, pv[ni][v]);
#pragma unroll
      for (int d = 1; d < 16; d <<= 1) mx = fmaxf(mx, __shfl_xor(mx, d, 32));
      float mnew  = fmaxf(mrun[v], mx);
      float alpha = __expf(mrun[v] - mnew);
      float rs = 0.0f;
#pragma unroll
      for (int ni = 0; ni < 4; ++ni) {
        float p = __expf(pv[ni][v] - mnew);
        pv[ni][v] = p;
        rs += p;
      }
#pragma unroll
      for (int d = 1; d < 16; d <<= 1) rs += __shfl_xor(rs, d, 32);
      lrun[v] = lrun[v] * alpha + rs;
      mrun[v] = mnew;
#pragma unroll
      for (int ni = 0; ni < 4; ++ni) o[ni][v] *= alpha;
    }

    // P: D-layout -> A-layout via wave-private LDS (same-wave DS is in-order)
#pragma unroll
    for (int ni = 0; ni < 4; ++ni)
#pragma unroll
      for (int v = 0; v < 8; ++v)
        Ps[(wave * 16 + v + 8 * hi) * KSTR + ni * 16 + ln] = (bf16)pv[ni][v];

    // O += P x V
#pragma unroll
    for (int kc = 0; kc < 2; ++kc) {
      int prow = wave * 16 + ln;
      bf16x8 plo = *(const bf16x8*)&Ps[prow * KSTR + kc * 32 + hi * 8];
      bf16x8 ph8 = *(const bf16x8*)&Ps[prow * KSTR + kc * 32 + 16 + hi * 8];
      bf16x16 pf = mk16(plo, ph8);
      int vrow = kc * 32 + hi * 8;  // B: lane<->N(hd), regs<->K(key)
#pragma unroll
      for (int ni = 0; ni < 4; ++ni) {
        bf16x16 vf;
#pragma unroll
        for (int e = 0; e < 16; ++e)
          vf[e] = Vs[(vrow + ((e & 8) << 1) + (e & 7)) * KSTR + ni * 16 + ln];
        o[ni] = wmma_bf16(pf, vf, o[ni]);
      }
    }
  }

  // Epilogue: normalize and write [B*T][H] bf16 (cols = h*64 + hd)
#pragma unroll
  for (int ni = 0; ni < 4; ++ni)
#pragma unroll
    for (int v = 0; v < 8; ++v) {
      int t   = qb * 64 + wave * 16 + v + 8 * hi;
      int col = h * 64 + ni * 16 + ln;
      float val = o[ni][v] / lrun[v];
      aout[((size_t)b * T_ + t) * H_ + col] = (bf16)val;
    }
}

// ---------------------------------------------------------------------------
// Kernel 3: out (fp32) = attn (bf16) @ Wout^T ; same tiling as kernel 1
// ---------------------------------------------------------------------------
__global__ void __launch_bounds__(256)
out_gemm_kernel(const bf16* __restrict__ X, const float* __restrict__ W,
                float* __restrict__ out) {
  __shared__ bf16 As[128 * ASTR];
  __shared__ bf16 Bt[32 * BSTR];

  const int tid  = threadIdx.x;
  const int lane = tid & 31;
  const int wave = tid >> 5;
  const int wm   = wave >> 1;
  const int wn   = wave & 1;
  const int hi   = lane >> 4;
  const int ln   = lane & 15;

  const int m0 = blockIdx.y * 128;
  const int n0 = blockIdx.x * 128;

  v8f acc[2][4];
#pragma unroll
  for (int i = 0; i < 2; ++i)
#pragma unroll
    for (int j = 0; j < 4; ++j) acc[i][j] = (v8f)0.0f;

  for (int kb = 0; kb < KDIM; kb += 32) {
    __syncthreads();
    // X tile already bf16: straight LDS copy
#pragma unroll
    for (int p = 0; p < 2; ++p) {
      int r = (tid >> 2) + p * 64;
      int c = (tid & 3) * 8;
      *(bf16x8*)&As[r * ASTR + c] =
          *(const bf16x8*)(X + (size_t)(m0 + r) * KDIM + kb + c);
    }
    // W tile fp32 -> transposed bf16
#pragma unroll
    for (int p = 0; p < 4; ++p) {
      int r = (tid >> 3) + p * 32;
      int c = (tid & 7) * 4;
      float4 f = *(const float4*)(W + (size_t)(n0 + r) * KDIM + kb + c);
      Bt[(c + 0) * BSTR + r] = (bf16)f.x;
      Bt[(c + 1) * BSTR + r] = (bf16)f.y;
      Bt[(c + 2) * BSTR + r] = (bf16)f.z;
      Bt[(c + 3) * BSTR + r] = (bf16)f.w;
    }
    if (kb + 32 < KDIM) {
      __builtin_prefetch(X + (size_t)(m0 + (tid >> 2)) * KDIM + kb + 32, 0, 1);
      __builtin_prefetch(W + (size_t)(n0 + (tid >> 3)) * KDIM + kb + 32, 0, 1);
    }
    __syncthreads();

    bf16x16 afrag[2], bfrag[4];
#pragma unroll
    for (int mi = 0; mi < 2; ++mi) {
      int row = wm * 32 + mi * 16 + ln;
      bf16x8 lo = *(const bf16x8*)&As[row * ASTR + hi * 8];
      bf16x8 h8 = *(const bf16x8*)&As[row * ASTR + 16 + hi * 8];
      afrag[mi] = mk16(lo, h8);
    }
#pragma unroll
    for (int ni = 0; ni < 4; ++ni) {
      int col = wn * 64 + ni * 16 + ln;
#pragma unroll
      for (int e = 0; e < 16; ++e)
        bfrag[ni][e] = Bt[(((e & 8) << 1) + hi * 8 + (e & 7)) * BSTR + col];
    }
#pragma unroll
    for (int mi = 0; mi < 2; ++mi)
#pragma unroll
      for (int ni = 0; ni < 4; ++ni)
        acc[mi][ni] = wmma_bf16(afrag[mi], bfrag[ni], acc[mi][ni]);
  }

#pragma unroll
  for (int mi = 0; mi < 2; ++mi)
#pragma unroll
    for (int ni = 0; ni < 4; ++ni)
#pragma unroll
      for (int v = 0; v < 8; ++v) {
        int m = m0 + wm * 32 + mi * 16 + v + 8 * hi;
        int n = n0 + wn * 64 + ni * 16 + ln;
        out[(size_t)m * H_ + n] = acc[mi][ni][v];
      }
}

// ---------------------------------------------------------------------------
extern "C" void kernel_launch(void* const* d_in, const int* in_sizes, int n_in,
                              void* d_out, int out_size, void* d_ws, size_t ws_size,
                              hipStream_t stream) {
  const float* x     = (const float*)d_in[0];  // [B,T,H]
  const float* w_qkv = (const float*)d_in[1];  // [3H,H]
  const float* w_out = (const float*)d_in[2];  // [H,H]
  float* out = (float*)d_out;                  // [B,T,H]

  bf16* ws   = (bf16*)d_ws;
  bf16* qkv  = ws;                                   // [3][B][NH][T][HD] bf16 (48 MiB)
  bf16* attn = ws + (size_t)3 * B_ * T_ * H_;        // [B*T][H] bf16 (16 MiB)

  dim3 g1(3 * H_ / 128, M_ / 128);  // (24, 64)
  qkv_gemm_kernel<<<g1, 256, 0, stream>>>(x, w_qkv, qkv);

  dim3 g2(T_ / 64, B_ * NH_);       // (32, 64)
  attn_kernel<<<g2, 128, 0, stream>>>(qkv, attn);

  dim3 g3(H_ / 128, M_ / 128);      // (8, 64)
  out_gemm_kernel<<<g3, 256, 0, stream>>>(attn, w_out, out);
}
